// LiveRec_55035710931236
// MI455X (gfx1250) — matrix operand order repaired and checked by reading, not compiled, not run on hardware
//
#include <hip/hip_runtime.h>
#include <math.h>

// Problem constants (B,S,K,A,T,NV,TOPK,H = 32,64,128,512,1024,100000,64,4)
#define KD    128
#define AQ    512
#define TOPKQ 64
#define HQ    4
#define DH    32
#define MTOK  2048        // B*S

// Padded LDS strides chosen for 16B vector alignment + bank spread
#define LDK   136         // bf16 row stride for 128-wide matrices (272B)
#define LDV   72          // bf16 row stride for Vt (144B)
#define LGS   68          // f32 row stride for logits (272B == LDK halves)

typedef __bf16 v16bf __attribute__((ext_vector_type(16)));
typedef __bf16 bf8v  __attribute__((ext_vector_type(8)));   // 16-byte LDS vector
typedef float  v8f   __attribute__((ext_vector_type(8)));

// LDS carve-out (dynamic shared), all offsets 16B aligned.  Total ~137KB ->
// 2 blocks (16 waves) resident per 320KB WGP.
#define SB_OFF 0u                           // S (bf16 64xLDK), reused as O
#define QB_OFF (SB_OFF + 64u * LDK * 2u)    // 17408
#define KB_OFF (QB_OFF + 64u * LDK * 2u)    // 34816
#define VT_OFF (KB_OFF + 64u * LDK * 2u)    // 52224: V^T [128][LDV] bf16
#define LG_OFF (VT_OFF + 128u * LDV * 2u)   // 70656: logits [2][64][LGS] f32,
                                            //        softmaxed bf16 att in place
#define WT_OFF (LG_OFF + 2u * 64u * LGS * 4u) // 105472: staged W^T [128][LDK]
#define SMEM_TOTAL (WT_OFF + 128u * LDK * 2u) // 140288

__device__ __forceinline__ v16bf concat8(bf8v a, bf8v b) {
  return __builtin_shufflevector(a, b, 0, 1, 2, 3, 4, 5, 6, 7,
                                 8, 9, 10, 11, 12, 13, 14, 15);
}

// A fragment (16x32 bf16): lane lo = row M; per-lane K = one contiguous run of 16 at 16*hi
__device__ __forceinline__ v16bf load_A_lds(const __bf16* base, int ld, int lane) {
  const int lo = lane & 15, hi = lane >> 4;
  const __bf16* row = base + lo * ld + 16 * hi;
  const bf8v r0 = *(const bf8v*)(row);
  const bf8v r1 = *(const bf8v*)(row + 8);
  return concat8(r0, r1);
}

// B fragment from a TRANSPOSED row-major source: logical B[k][n] = src[n][k]
// -> per lane one contiguous run of 16 halves at (row lo, col 16*hi)
__device__ __forceinline__ v16bf load_BT_lds(const __bf16* base, int ld, int lane) {
  const int lo = lane & 15, hi = lane >> 4;
  const __bf16* row = base + lo * ld + 16 * hi;
  const bf8v r0 = *(const bf8v*)(row);
  const bf8v r1 = *(const bf8v*)(row + 8);
  return concat8(r0, r1);
}

// D (16x16 f32): VGPR r -> M = r + 8*hi, N = lo  (row-major bf16 store)
__device__ __forceinline__ void store_D_bf16(__bf16* base, int ld, int lane, v8f d) {
  const int lo = lane & 15, hi = lane >> 4;
#pragma unroll
  for (int r = 0; r < 8; ++r) base[(r + 8 * hi) * ld + lo] = (__bf16)d[r];
}

// D stored TRANSPOSED: dst[N][M] -> per-lane 8 contiguous halves, one 16B store
__device__ __forceinline__ void store_D_bf16_T(__bf16* dst, int ld, int m0, int n0,
                                               int lane, v8f d) {
  const int lo = lane & 15, hi = lane >> 4;
  bf8v v;
#pragma unroll
  for (int r = 0; r < 8; ++r) v[r] = (__bf16)d[r];
  *(bf8v*)(dst + (size_t)(n0 + lo) * ld + m0 + 8 * hi) = v;
}

__device__ __forceinline__ void store_D_f32(float* base, int ld, int lane, v8f d,
                                            float s) {
  const int lo = lane & 15, hi = lane >> 4;
#pragma unroll
  for (int r = 0; r < 8; ++r) base[(r + 8 * hi) * ld + lo] = d[r] * s;
}

__device__ __forceinline__ v8f wmma_bf16(v16bf a, v16bf b, v8f c) {
  return __builtin_amdgcn_wmma_f32_16x16x32_bf16(false, a, false, b, (short)0, c,
                                                 false, false);
}

// Stage W^T into LDS as bf16: Wt[n][k] = W[k][n]; coalesced float4 global reads.
__device__ __forceinline__ void stage_weight_T(const float* __restrict__ W,
                                               __bf16* Wt, int tid) {
#pragma unroll 1
  for (int it = 0; it < 16; ++it) {
    const int idx4 = tid + it * 256;          // 4096 float4 total
    const int k = idx4 >> 5;                  // row of W
    const int n0 = (idx4 & 31) * 4;           // col of W
    const float4 v = ((const float4*)(W + (size_t)k * KD))[idx4 & 31];
    __bf16* col = Wt + k;
    col[(size_t)(n0 + 0) * LDK] = (__bf16)v.x;
    col[(size_t)(n0 + 1) * LDK] = (__bf16)v.y;
    col[(size_t)(n0 + 2) * LDK] = (__bf16)v.z;
    col[(size_t)(n0 + 3) * LDK] = (__bf16)v.w;
  }
}

// 16x32 output slab: shared A fragment, two B columns, software-pipelined k-steps
// A: 64xKD source (ld LDK), B: transposed weight-style source (ld LDK), 4 k-steps
__device__ __forceinline__ void gemm_slab_k128(const __bf16* Abase,
                                               const __bf16* B0, const __bf16* B1,
                                               int lane, v8f& acc0, v8f& acc1) {
  v16bf a  = load_A_lds(Abase, LDK, lane);
  v16bf b0 = load_BT_lds(B0, LDK, lane);
  v16bf b1 = load_BT_lds(B1, LDK, lane);
#pragma unroll
  for (int kt = 0; kt < 4; ++kt) {
    v16bf an = a, b0n = b0, b1n = b1;
    if (kt < 3) {                      // prefetch next k-step before the WMMAs
      an  = load_A_lds(Abase + (kt + 1) * 32, LDK, lane);
      b0n = load_BT_lds(B0 + (kt + 1) * 32, LDK, lane);
      b1n = load_BT_lds(B1 + (kt + 1) * 32, LDK, lane);
    }
    acc0 = wmma_bf16(a, b0, acc0);
    acc1 = wmma_bf16(a, b1, acc1);
    a = an; b0 = b0n; b1 = b1n;
  }
}

// ---------------------------------------------------------------------------
// Kernel 1: per-token scores (512 dot-128), bitonic sort (score desc, index asc
// == jax.lax.top_k tie-break), masked batch_inds, gather winning 64 rows (f32)
// into the output buffer for kernel 2.
// ---------------------------------------------------------------------------
__global__ __launch_bounds__(256) void topk_gather_kernel(
    const int* __restrict__ inputs, const int* __restrict__ xtsy,
    const float* __restrict__ feats, const float* __restrict__ item_emb,
    const int* __restrict__ av_tens, float* __restrict__ out_seqs,
    int* __restrict__ batch_inds) {
  __shared__ float fsh[KD];
  __shared__ float sc[AQ];
  __shared__ int   si[AQ];
  const int m   = blockIdx.x;
  const int tid = threadIdx.x;
  const int* avrow = av_tens + (size_t)xtsy[m] * AQ;
  if (tid < KD) fsh[tid] = feats[(size_t)m * KD + tid];
  __syncthreads();

  const float4* f4 = (const float4*)fsh;
  for (int a = tid; a < AQ; a += 256) {
    const int idx = avrow[a];
    const float4* e4 = (const float4*)(item_emb + (size_t)idx * KD);
    float s = 0.f;
#pragma unroll 8
    for (int i = 0; i < KD / 4; ++i) {
      const float4 e = e4[i];
      const float4 f = f4[i];
      s += e.x * f.x + e.y * f.y + e.z * f.z + e.w * f.w;
    }
    sc[a] = s;
    si[a] = a;
  }

  for (int ksz = 2; ksz <= AQ; ksz <<= 1) {
    for (int jsz = ksz >> 1; jsz > 0; jsz >>= 1) {
      __syncthreads();
      for (int i = tid; i < AQ; i += 256) {
        const int ixj = i ^ jsz;
        if (ixj > i) {
          const float sa = sc[i], sb = sc[ixj];
          const int   ia = si[i], ib = si[ixj];
          const bool bpref = (sb > sa) || (sb == sa && ib < ia);
          const bool desc  = ((i & ksz) == 0);
          if (desc ? bpref : !bpref) {
            sc[i] = sb; sc[ixj] = sa;
            si[i] = ib; si[ixj] = ia;
          }
        }
      }
    }
  }
  __syncthreads();

  const bool active = (inputs[m] != 0);
  if (tid < TOPKQ) batch_inds[(size_t)m * TOPKQ + tid] = active ? si[tid] : 0;
  if (active) {
    float* outp = out_seqs + (size_t)m * TOPKQ * KD;
    for (int e = tid; e < TOPKQ * KD; e += 256) {
      const int t = e >> 7;
      const int k = e & (KD - 1);
      outp[e] = item_emb[(size_t)avrow[si[t]] * KD + k];
    }
  }
}

// ---------------------------------------------------------------------------
// Kernel 2: one block (8 waves) per token. All GEMMs via v_wmma bf16 in
// 16x32 slabs (shared A fragment, pipelined k-steps); weights staged
// transposed in LDS; V kept transposed; heads in pairs with in-place bf16 att;
// in-place residual against the f32 seqs kernel 1 left in d_out.
// ---------------------------------------------------------------------------
__global__ __launch_bounds__(256) void mha_wmma_kernel(
    const int* __restrict__ inputs, const float* __restrict__ Wq,
    const float* __restrict__ Wk, const float* __restrict__ Wv,
    const float* __restrict__ Wo, float* __restrict__ out) {
  extern __shared__ char smem[];
  __bf16* Sb = (__bf16*)(smem + SB_OFF);   // S, later reused as O
  __bf16* Qb = (__bf16*)(smem + QB_OFF);
  __bf16* Kb = (__bf16*)(smem + KB_OFF);
  __bf16* Vt = (__bf16*)(smem + VT_OFF);   // V transposed [128][LDV]
  float*  Lg = (float*)(smem + LG_OFF);    // logits [2][64][LGS] (att in place)
  __bf16* Wt = (__bf16*)(smem + WT_OFF);   // staged weight^T [128][LDK]

  const int m = blockIdx.x;
  const int tid = threadIdx.x;
  const int lane = tid & 31;
  const int wv = tid >> 5;                 // wave id 0..7
  float* outp = out + (size_t)m * TOPKQ * KD;

  if (inputs[m] == 0) {
    float4* o4 = (float4*)outp;
    const float4 z = {0.f, 0.f, 0.f, 0.f};
    for (int i = tid; i < TOPKQ * KD / 4; i += 256) o4[i] = z;
    return;
  }

  // seqs (f32 in d_out) -> bf16 LDS (padded rows)
  {
    const float4* s4 = (const float4*)outp;
    for (int i = tid; i < TOPKQ * KD / 4; i += 256) {
      const float4 v = s4[i];
      const int row = (i * 4) >> 7;
      const int col = (i * 4) & (KD - 1);
      __bf16* d = Sb + row * LDK + col;
      d[0] = (__bf16)v.x; d[1] = (__bf16)v.y;
      d[2] = (__bf16)v.z; d[3] = (__bf16)v.w;
    }
  }

  // ---- Q = S@Wq, K = S@Wk : 16 slab-pairs / 8 waves, dst picked per weight
#pragma unroll 1
  for (int w = 0; w < 2; ++w) {
    const float* W = w ? Wk : Wq;
    __bf16* Dst    = w ? Kb : Qb;
    __syncthreads();
    stage_weight_T(W, Wt, tid);
    __syncthreads();
#pragma unroll 1
    for (int j = 0; j < 2; ++j) {
      const int pi = wv + 8 * j;           // 0..15 slab index
      const int mt = pi >> 2;
      const int n0 = (pi & 3) * 2;
      v8f acc0 = {}, acc1 = {};
      gemm_slab_k128(Sb + mt * 16 * LDK, Wt + (n0 * 16) * LDK,
                     Wt + ((n0 + 1) * 16) * LDK, lane, acc0, acc1);
      store_D_bf16(Dst + mt * 16 * LDK + n0 * 16, LDK, lane, acc0);
      store_D_bf16(Dst + mt * 16 * LDK + (n0 + 1) * 16, LDK, lane, acc1);
    }
  }

  // ---- V^T = (S@Wv)^T : transposed D store, one 16B LDS store per fragment
  __syncthreads();
  stage_weight_T(Wv, Wt, tid);
  __syncthreads();
#pragma unroll 1
  for (int j = 0; j < 2; ++j) {
    const int pi = wv + 8 * j;
    const int mt = pi >> 2;
    const int n0 = (pi & 3) * 2;
    v8f acc0 = {}, acc1 = {};
    gemm_slab_k128(Sb + mt * 16 * LDK, Wt + (n0 * 16) * LDK,
                   Wt + ((n0 + 1) * 16) * LDK, lane, acc0, acc1);
    store_D_bf16_T(Vt, LDV, mt * 16, n0 * 16, lane, acc0);
    store_D_bf16_T(Vt, LDV, mt * 16, (n0 + 1) * 16, lane, acc1);
  }
  __syncthreads();

  // ---- attention, two heads at a time (halves the logits LDS footprint)
  const float inv_sqrt_d = 0.17677669529663687f;  // 1/sqrt(32)
#pragma unroll 1
  for (int pair = 0; pair < 2; ++pair) {
    // logits: 16 slab-pairs (shared Q fragment across two K tiles) / 8 waves
#pragma unroll 1
    for (int j = 0; j < 2; ++j) {
      const int pi = wv + 8 * j;           // 0..15
      const int p = pi >> 3, qt = (pi >> 1) & 3, k2 = (pi & 1) * 2;
      const int h = pair * 2 + p;
      v16bf a  = load_A_lds(Qb + qt * 16 * LDK + h * DH, LDK, lane);
      v16bf b0 = load_BT_lds(Kb + (k2 * 16) * LDK + h * DH, LDK, lane);
      v16bf b1 = load_BT_lds(Kb + ((k2 + 1) * 16) * LDK + h * DH, LDK, lane);
      v8f acc0 = {}, acc1 = {};
      acc0 = wmma_bf16(a, b0, acc0);
      acc1 = wmma_bf16(a, b1, acc1);
      float* Lbase = Lg + (p * 64 + qt * 16) * LGS;
      store_D_f32(Lbase + k2 * 16, LGS, lane, acc0, inv_sqrt_d);
      store_D_f32(Lbase + (k2 + 1) * 16, LGS, lane, acc1, inv_sqrt_d);
    }
    __syncthreads();

    // softmax: one (head,row) per thread (128 rows); bf16 att written in place
    // over the f32 row (ascending overwrite touches only already-read bytes)
    if (tid < 128) {
      const int p = tid >> 6, r = tid & 63;
      float* Lrow = Lg + (p * 64 + r) * LGS;
      float mx = -3.0e38f;
      for (int c = 0; c < TOPKQ; ++c) mx = fmaxf(mx, Lrow[c]);
      float sum = 0.f;
      for (int c = 0; c < TOPKQ; ++c) {
        const float e = __expf(Lrow[c] - mx);
        Lrow[c] = e;
        sum += e;
      }
      const float inv = 1.f / sum;
      __bf16* Arow = (__bf16*)Lrow;
      for (int c = 0; c < TOPKQ; ++c) Arow[c] = (__bf16)(Lrow[c] * inv);
    }
    __syncthreads();

    // O = att @ V: one (head,qt) slab per wave, shared att fragment over the
    // two 16-wide V column tiles; att rows stride LGS f32 == LDK halves
    {
      const __bf16* AttB = (const __bf16*)Lg;
      const int p = wv >> 2, qt = wv & 3;
      const int h = pair * 2 + p;
      const __bf16* Abase = AttB + (p * 64 + qt * 16) * LDK;
      const __bf16* B0 = Vt + (h * DH) * LDV;
      const __bf16* B1 = Vt + (h * DH + 16) * LDV;
      v8f acc0 = {}, acc1 = {};
#pragma unroll
      for (int kt = 0; kt < 2; ++kt) {
        v16bf a  = load_A_lds(Abase + kt * 32, LDK, lane);
        v16bf b0 = load_BT_lds(B0 + kt * 32, LDV, lane);
        v16bf b1 = load_BT_lds(B1 + kt * 32, LDV, lane);
        acc0 = wmma_bf16(a, b0, acc0);
        acc1 = wmma_bf16(a, b1, acc1);
      }
      store_D_bf16(Sb + qt * 16 * LDK + h * DH, LDK, lane, acc0);       // O->Sb
      store_D_bf16(Sb + qt * 16 * LDK + h * DH + 16, LDK, lane, acc1);
    }
    __syncthreads();
  }

  // ---- out = seqs(f32, intact in global) + O @ Wo
  stage_weight_T(Wo, Wt, tid);
  __syncthreads();
#pragma unroll 1
  for (int j = 0; j < 2; ++j) {
    const int pi = wv + 8 * j;
    const int mt = pi >> 2;
    const int n0 = (pi & 3) * 2;
    v8f acc0 = {}, acc1 = {};
    gemm_slab_k128(Sb + mt * 16 * LDK, Wt + (n0 * 16) * LDK,
                   Wt + ((n0 + 1) * 16) * LDK, lane, acc0, acc1);
    const int lo = lane & 15, hi = lane >> 4;
    float* ob0 = outp + (size_t)(mt * 16) * KD + n0 * 16;
    float* ob1 = ob0 + 16;
#pragma unroll
    for (int r = 0; r < 8; ++r) {
      const int off = (r + 8 * hi) * KD + lo;
      ob0[off] = ob0[off] + acc0[r];
      ob1[off] = ob1[off] + acc1[r];
    }
  }
}

// ---------------------------------------------------------------------------
extern "C" void kernel_launch(void* const* d_in, const int* in_sizes, int n_in,
                              void* d_out, int out_size, void* d_ws, size_t ws_size,
                              hipStream_t stream) {
  (void)in_sizes; (void)n_in; (void)out_size; (void)d_ws; (void)ws_size;
  const int*   inputs   = (const int*)d_in[0];
  const int*   xtsy     = (const int*)d_in[1];
  const float* feats    = (const float*)d_in[2];
  const float* item_emb = (const float*)d_in[3];
  const int*   av_tens  = (const int*)d_in[4];
  const float* Wq       = (const float*)d_in[5];
  const float* Wk       = (const float*)d_in[6];
  const float* Wv       = (const float*)d_in[7];
  const float* Wo       = (const float*)d_in[8];

  float* out = (float*)d_out;
  int* binds = (int*)(out + (size_t)MTOK * TOPKQ * KD);

  topk_gather_kernel<<<MTOK, 256, 0, stream>>>(inputs, xtsy, feats, item_emb,
                                               av_tens, out, binds);
  mha_wmma_kernel<<<MTOK, 256, SMEM_TOTAL, stream>>>(inputs, Wq, Wk, Wv, Wo, out);
}
